// HSMM_66571993088832
// MI455X (gfx1250) — matrix operand order repaired and checked by reading, not compiled, not run on hardware
//
#include <hip/hip_runtime.h>
#include <hip/hip_bf16.h>

#define Bc 32
#define Tc 4096
#define Dc 64
#define Sc 64
#define Hc 128
#define NBT (Bc * Tc * Sc) /* 8388608 */
#define EPSc 1e-10f

typedef _Float16 v16h __attribute__((ext_vector_type(16)));
typedef _Float16 v8h  __attribute__((ext_vector_type(8)));
typedef float    v8f  __attribute__((ext_vector_type(8)));

// ---------------- wave32 cross-lane helpers ----------------
__device__ __forceinline__ float wredmax(float v) {
    v = fmaxf(v, __shfl_xor(v, 16, 32));
    v = fmaxf(v, __shfl_xor(v, 8, 32));
    v = fmaxf(v, __shfl_xor(v, 4, 32));
    v = fmaxf(v, __shfl_xor(v, 2, 32));
    v = fmaxf(v, __shfl_xor(v, 1, 32));
    return v;
}
__device__ __forceinline__ float wredsum(float v) {
    v += __shfl_xor(v, 16, 32);
    v += __shfl_xor(v, 8, 32);
    v += __shfl_xor(v, 4, 32);
    v += __shfl_xor(v, 2, 32);
    v += __shfl_xor(v, 1, 32);
    return v;
}
// uniform broadcast from a constant lane: lowers to v_readlane_b32 (no LDS pipe)
__device__ __forceinline__ float lane_bcast(float v, int lane) {
    return __int_as_float(__builtin_amdgcn_readlane(__float_as_int(v), lane));
}

// ---------------- WMMA helpers ----------------
__device__ __forceinline__ v8f wmma16(v16h a, v16h b, v8f c) {
    // v_wmma_f32_16x16x32_f16, D = A*B + C
    return __builtin_amdgcn_wmma_f32_16x16x32_f16(false, a, false, b, (short)0, c,
                                                  false, false);
}

// B tile pre-swizzled in global memory: [tile][lane][16 halves], 32B/lane.
__device__ __forceinline__ v16h load_b_tile(const _Float16* __restrict__ w, int tile,
                                            int lane) {
    return *(const v16h*)(w + ((size_t)tile * 32 + lane) * 16);
}

// A tile (16x32 f16) from a row-major f32 row of 64 elements (layer 1 input).
// ISA 16-bit A layout: lane<16 holds K = b0..b0+7 and b0+16..b0+23 (b0=k*32+khi*8).
__device__ __forceinline__ v16h load_a_f32row(const float* __restrict__ row, int khi,
                                              int k) {
    int b0 = k * 32 + khi * 8;
    float4 f0 = *(const float4*)(row + b0);
    float4 f1 = *(const float4*)(row + b0 + 4);
    float4 f2 = *(const float4*)(row + b0 + 16);
    float4 f3 = *(const float4*)(row + b0 + 20);
    v16h a;
    a[0] = (_Float16)f0.x;  a[1] = (_Float16)f0.y;
    a[2] = (_Float16)f0.z;  a[3] = (_Float16)f0.w;
    a[4] = (_Float16)f1.x;  a[5] = (_Float16)f1.y;
    a[6] = (_Float16)f1.z;  a[7] = (_Float16)f1.w;
    a[8] = (_Float16)f2.x;  a[9] = (_Float16)f2.y;
    a[10] = (_Float16)f2.z; a[11] = (_Float16)f2.w;
    a[12] = (_Float16)f3.x; a[13] = (_Float16)f3.y;
    a[14] = (_Float16)f3.z; a[15] = (_Float16)f3.w;
    return a;
}

// A tile from a row-major 16x128 f16 LDS slice.
__device__ __forceinline__ v16h load_a_lds(const _Float16* __restrict__ base, int M,
                                           int khi, int k) {
    int b0 = k * 32 + khi * 8;
    v8h lo = *(const v8h*)(base + M * 128 + b0);
    v8h hi = *(const v8h*)(base + M * 128 + b0 + 16);
    v16h a;
#pragma unroll
    for (int i = 0; i < 8; ++i) { a[i] = lo[i]; a[8 + i] = hi[i]; }
    return a;
}

// ---------------- weight swizzle (f32 -> f16 WMMA B-operand order) -------------
__global__ void swz_weights_kernel(const float* __restrict__ W1,
                                   const float* __restrict__ W2,
                                   const float* __restrict__ W3,
                                   _Float16* __restrict__ w1h,
                                   _Float16* __restrict__ w2h,
                                   _Float16* __restrict__ w3h) {
    int idx = blockIdx.x * blockDim.x + threadIdx.x;
    const float* src; _Float16* dst; int N;
    if (idx < 8192)        { src = W1; dst = w1h; N = 128; }
    else if (idx < 24576)  { idx -= 8192;  src = W2; dst = w2h; N = 128; }
    else if (idx < 32768)  { idx -= 24576; src = W3; dst = w3h; N = 64;  }
    else return;
    int tile = idx >> 9, within = idx & 511;
    int lane = within >> 4, h = within & 15;
    int ntiles = N >> 4;
    int kt = tile / ntiles, nt = tile - kt * ntiles;
    int krow = kt * 32 + ((lane >> 4) << 3) + (h & 7) + ((h >> 3) << 4);
    int ncol = nt * 16 + (lane & 15);
    dst[tile * 512 + lane * 16 + h] = (_Float16)src[krow * N + ncol];
}

// ---------------- transition matrix: masked row softmax + eps ----------------
__global__ __launch_bounds__(64) void prep_trans_kernel(const float* __restrict__ ltm,
                                                        float* __restrict__ trans,
                                                        float* __restrict__ transT) {
    int srow = blockIdx.x, j = threadIdx.x;
    int lane = j & 31, wave = j >> 5;
    __shared__ float r1[2], r2[2];
    float v = (j == srow) ? -1e10f : ltm[srow * Sc + j];
    float m = wredmax(v);
    if (lane == 0) r1[wave] = m;
    __syncthreads();
    m = fmaxf(r1[0], r1[1]);
    float e = __expf(v - m);
    float ss = wredsum(e);
    if (lane == 0) r2[wave] = ss;
    __syncthreads();
    float p = e / (r2[0] + r2[1]) + EPSc;  // exp(log(p+eps)) used in LSE trick
    trans[srow * Sc + j] = p;
    transT[j * Sc + srow] = p;
}

// ---------------- log-init normalization + expected durations ----------------
__global__ __launch_bounds__(64) void prep_misc_kernel(const float* __restrict__ lip,
                                                       const float* __restrict__ log_r,
                                                       const float* __restrict__ logit_p,
                                                       float* __restrict__ linit,
                                                       float* __restrict__ out_dur) {
    int s = threadIdx.x;
    int lane = s & 31, wave = s >> 5;
    __shared__ float r1[2], r2[2];
    float rr = __expf(log_r[s]);
    float pp = 1.0f / (1.0f + __expf(-logit_p[s]));
    out_dur[s] = rr * (1.0f - pp) / pp;
    float v = lip[s];
    float m = wredmax(v);
    if (lane == 0) r1[wave] = m;
    __syncthreads();
    m = fmaxf(r1[0], r1[1]);
    float e = __expf(v - m);
    float ss = wredsum(e);
    if (lane == 0) r2[wave] = ss;
    __syncthreads();
    linit[s] = v - m - __logf(r2[0] + r2[1]);
}

// ---------------- emission MLP + log-softmax (WMMA) ----------------
// Block: 128 threads = 4 waves; wave handles 16 rows of (B*T). 2048 blocks.
__global__ __launch_bounds__(128) void mlp_kernel(const float* __restrict__ obs,
                                                  const _Float16* __restrict__ w1h,
                                                  const _Float16* __restrict__ w2h,
                                                  const _Float16* __restrict__ w3h,
                                                  const float* __restrict__ b1,
                                                  const float* __restrict__ b2,
                                                  const float* __restrict__ b3,
                                                  float* __restrict__ logem) {
    __shared__ _Float16 sH1[4][16][128];  // 16 KB (reused as f32 logits)
    __shared__ _Float16 sH2[4][16][128];  // 16 KB
    const int tid = threadIdx.x, wave = tid >> 5, lane = tid & 31;
    const int ln15 = lane & 15, khi = (lane >> 4) & 1;
    const int row0 = blockIdx.x * 64;
    const int Mrow = row0 + wave * 16 + ln15;
    const float* orow = obs + (size_t)Mrow * Dc;

    // hint: pull hot (shared) weights toward the WGP
    __builtin_prefetch(w2h + (size_t)tid * 128, 0, 0);

    // ---- layer 1: X(16x64) @ W1(64x128) ----
    v16h a0 = load_a_f32row(orow, khi, 0);
    v16h a1 = load_a_f32row(orow, khi, 1);
    _Float16* h1 = &sH1[wave][0][0];
#pragma unroll
    for (int j = 0; j < 8; ++j) {
        v8f acc = {};
        acc = wmma16(a0, load_b_tile(w1h, 0 * 8 + j, lane), acc);
        acc = wmma16(a1, load_b_tile(w1h, 1 * 8 + j, lane), acc);
        float bias = b1[j * 16 + ln15];
#pragma unroll
        for (int r = 0; r < 8; ++r) {
            float v = acc[r] + bias;
            v = v > 0.0f ? v : 0.0f;
            int M = (lane < 16) ? r : (r + 8);
            h1[M * 128 + j * 16 + ln15] = (_Float16)v;
        }
    }
    __syncthreads();

    // ---- layer 2: H1(16x128) @ W2(128x128) ----
    v16h a2[4];
#pragma unroll
    for (int k = 0; k < 4; ++k) a2[k] = load_a_lds(h1, ln15, khi, k);
    _Float16* h2 = &sH2[wave][0][0];
#pragma unroll
    for (int j = 0; j < 8; ++j) {
        v8f acc = {};
#pragma unroll
        for (int k = 0; k < 4; ++k)
            acc = wmma16(a2[k], load_b_tile(w2h, k * 8 + j, lane), acc);
        float bias = b2[j * 16 + ln15];
#pragma unroll
        for (int r = 0; r < 8; ++r) {
            float v = acc[r] + bias;
            v = v > 0.0f ? v : 0.0f;
            int M = (lane < 16) ? r : (r + 8);
            h2[M * 128 + j * 16 + ln15] = (_Float16)v;
        }
    }
    __syncthreads();

    // ---- layer 3: H2(16x128) @ W3(128x64) -> logits (f32, reuse sH1 space) ----
    v16h a3[4];
#pragma unroll
    for (int k = 0; k < 4; ++k) a3[k] = load_a_lds(h2, ln15, khi, k);
    float* slog = (float*)&sH1[0][0][0];  // [64 rows][64 cols] f32 == 16 KB
#pragma unroll
    for (int j = 0; j < 4; ++j) {
        v8f acc = {};
#pragma unroll
        for (int k = 0; k < 4; ++k)
            acc = wmma16(a3[k], load_b_tile(w3h, k * 4 + j, lane), acc);
        float bias = b3[j * 16 + ln15];
#pragma unroll
        for (int r = 0; r < 8; ++r) {
            int M = (lane < 16) ? r : (r + 8);
            slog[(wave * 16 + M) * 64 + j * 16 + ln15] = acc[r] + bias;
        }
    }
    __syncthreads();

    // ---- softmax + log(p+eps): 2 threads per row, halves joined by shfl_xor(1) ----
    int row = tid >> 1, halfsel = tid & 1;
    const float* xr = slog + row * 64 + halfsel * 32;
    float m = -3.4e38f;
#pragma unroll 8
    for (int i = 0; i < 32; ++i) m = fmaxf(m, xr[i]);
    m = fmaxf(m, __shfl_xor(m, 1, 32));
    float Z = 0.0f;
#pragma unroll 8
    for (int i = 0; i < 32; ++i) Z += __expf(xr[i] - m);
    Z += __shfl_xor(Z, 1, 32);
    float invZ = 1.0f / Z;
    float* outr = logem + (size_t)(row0 + row) * Sc + halfsel * 32;
#pragma unroll 8
    for (int i = 0; i < 32; ++i)
        outr[i] = __logf(__expf(xr[i] - m) * invZ + EPSc);
}

// ---------------- forward recursion: one wave32 per batch ----------------
// Thread i owns states 2i, 2i+1. Transition matrix fully register-resident:
// tr[2*s0+c] = (p+eps)[s0][2i+c]. Cross-lane e-broadcast via v_readlane
// (uniform constant lane). Zero barriers, zero LDS in the 4096-step chain.
__global__ __launch_bounds__(32) void forward_kernel(const float* __restrict__ logem,
                                                     const float* __restrict__ trans,
                                                     const float* __restrict__ linit,
                                                     float* __restrict__ alpha,
                                                     float* __restrict__ out_ll) {
    const int b = blockIdx.x, i = threadIdx.x;
    const float* em_b = logem + (size_t)b * Tc * Sc;
    float* al_b = alpha + (size_t)b * Tc * Sc;

    float tr[2 * Sc];
#pragma unroll
    for (int s0 = 0; s0 < Sc; ++s0) {
        float2 p = *(const float2*)(trans + s0 * Sc + 2 * i);
        tr[2 * s0] = p.x;
        tr[2 * s0 + 1] = p.y;
    }
    float2 li = *(const float2*)(linit + 2 * i);
    float2 e0v = *(const float2*)(em_b + 2 * i);
    float la0 = li.x + e0v.x;
    float la1 = li.y + e0v.y;
    *(float2*)(al_b + 2 * i) = make_float2(la0, la1);

    for (int t = 1; t < Tc; ++t) {
        __builtin_prefetch(em_b + (size_t)((t + 8 < Tc) ? t + 8 : t) * Sc + 2 * i, 0, 0);
        float m = wredmax(fmaxf(la0, la1));
        float e0 = __expf(la0 - m);
        float e1 = __expf(la1 - m);
        float dot0 = 0.0f, dot1 = 0.0f;
#pragma unroll
        for (int s0 = 0; s0 < Sc; ++s0) {
            float ev = lane_bcast((s0 & 1) ? e1 : e0, s0 >> 1);  // v_readlane
            dot0 = fmaf(ev, tr[2 * s0], dot0);
            dot1 = fmaf(ev, tr[2 * s0 + 1], dot1);
        }
        float2 em2 = *(const float2*)(em_b + (size_t)t * Sc + 2 * i);
        la0 = m + __logf(dot0) + em2.x;
        la1 = m + __logf(dot1) + em2.y;
        *(float2*)(al_b + (size_t)t * Sc + 2 * i) = make_float2(la0, la1);
    }
    // log-likelihood = logsumexp(la)
    float m = wredmax(fmaxf(la0, la1));
    float Z = wredsum(__expf(la0 - m) + __expf(la1 - m));
    if (i == 0) out_ll[b] = m + __logf(Z);
}

// --------- backward recursion fused with posterior/marginals: 1 wave32/batch ------
__global__ __launch_bounds__(32) void backward_kernel(const float* __restrict__ logem,
                                                      const float* __restrict__ transT,
                                                      const float* __restrict__ alpha,
                                                      float* __restrict__ probs,
                                                      float* __restrict__ marg) {
    const int b = blockIdx.x, i = threadIdx.x;
    const float* em_b = logem + (size_t)b * Tc * Sc;
    const float* al_b = alpha + (size_t)b * Tc * Sc;
    float* pr_b = probs + (size_t)b * Tc * Sc;

    // tb[2*s0+c] = (p+eps)[2i+c][s0] (i.e. transT[s0][2i+c])
    float tb[2 * Sc];
#pragma unroll
    for (int s0 = 0; s0 < Sc; ++s0) {
        float2 p = *(const float2*)(transT + s0 * Sc + 2 * i);
        tb[2 * s0] = p.x;
        tb[2 * s0 + 1] = p.y;
    }

    float lb0 = 0.0f, lb1 = 0.0f, mac0 = 0.0f, mac1 = 0.0f;
    {   // t = T-1 posterior (beta = 0)
        float2 g2 = *(const float2*)(al_b + (size_t)(Tc - 1) * Sc + 2 * i);
        float m = wredmax(fmaxf(g2.x, g2.y));
        float Z = wredsum(__expf(g2.x - m) + __expf(g2.y - m));
        float lse = m + __logf(Z);
        float p0 = __expf(g2.x - lse), p1 = __expf(g2.y - lse);
        *(float2*)(pr_b + (size_t)(Tc - 1) * Sc + 2 * i) = make_float2(p0, p1);
        mac0 = p0; mac1 = p1;
    }
    for (int t = Tc - 2; t >= 0; --t) {
        __builtin_prefetch(em_b + (size_t)((t >= 8) ? t - 7 : t) * Sc + 2 * i, 0, 0);
        // beta step: lb[s] = m2 + log( sum_{s'} (p+eps)[s][s'] * exp(w[s'] - m2) )
        float2 em2 = *(const float2*)(em_b + (size_t)(t + 1) * Sc + 2 * i);
        float w0 = em2.x + lb0;
        float w1 = em2.y + lb1;
        float m2 = wredmax(fmaxf(w0, w1));
        float e0 = __expf(w0 - m2);
        float e1 = __expf(w1 - m2);
        float dot0 = 0.0f, dot1 = 0.0f;
#pragma unroll
        for (int s0 = 0; s0 < Sc; ++s0) {
            float ev = lane_bcast((s0 & 1) ? e1 : e0, s0 >> 1);  // v_readlane
            dot0 = fmaf(ev, tb[2 * s0], dot0);
            dot1 = fmaf(ev, tb[2 * s0 + 1], dot1);
        }
        lb0 = m2 + __logf(dot0);
        lb1 = m2 + __logf(dot1);
        // posterior at t
        float2 a2 = *(const float2*)(al_b + (size_t)t * Sc + 2 * i);
        float g0 = a2.x + lb0;
        float g1 = a2.y + lb1;
        float m = wredmax(fmaxf(g0, g1));
        float Z = wredsum(__expf(g0 - m) + __expf(g1 - m));
        float lse = m + __logf(Z);
        float p0 = __expf(g0 - lse), p1 = __expf(g1 - lse);
        *(float2*)(pr_b + (size_t)t * Sc + 2 * i) = make_float2(p0, p1);
        mac0 += p0; mac1 += p1;
    }
    *(float2*)(marg + b * Sc + 2 * i) =
        make_float2(mac0 * (1.0f / (float)Tc), mac1 * (1.0f / (float)Tc));
}

// ---------------- host launcher ----------------
extern "C" void kernel_launch(void* const* d_in, const int* in_sizes, int n_in,
                              void* d_out, int out_size, void* d_ws, size_t ws_size,
                              hipStream_t stream) {
    (void)in_sizes; (void)n_in; (void)out_size; (void)ws_size;
    const float* obs     = (const float*)d_in[0];
    const float* W1      = (const float*)d_in[1];
    const float* b1      = (const float*)d_in[2];
    const float* W2      = (const float*)d_in[3];
    const float* b2      = (const float*)d_in[4];
    const float* W3      = (const float*)d_in[5];
    const float* b3      = (const float*)d_in[6];
    const float* ltm     = (const float*)d_in[7];
    const float* lip     = (const float*)d_in[8];
    const float* log_r   = (const float*)d_in[9];
    const float* logit_p = (const float*)d_in[10];

    float* out   = (float*)d_out;
    float* probs = out;                     // (B,T,S)
    float* marg  = out + NBT;               // (B,S)
    float* dur   = out + NBT + Bc * Sc;     // (S,)
    float* ll    = out + NBT + Bc * Sc + Sc;// (B,)

    float* ws     = (float*)d_ws;
    float* logem  = ws;                  // NBT floats
    float* alpha  = ws + (size_t)NBT;    // NBT floats
    float* trans  = ws + (size_t)2 * NBT;        // 4096
    float* transT = trans + Sc * Sc;             // 4096
    float* linit  = transT + Sc * Sc;            // 64
    _Float16* w1h = (_Float16*)(ws + (size_t)2 * NBT + 2 * Sc * Sc + Sc); // 32B-aligned
    _Float16* w2h = w1h + 8192;
    _Float16* w3h = w2h + 16384;

    swz_weights_kernel<<<128, 256, 0, stream>>>(W1, W2, W3, w1h, w2h, w3h);
    prep_trans_kernel<<<Sc, 64, 0, stream>>>(ltm, trans, transT);
    prep_misc_kernel<<<1, 64, 0, stream>>>(lip, log_r, logit_p, linit, dur);
    mlp_kernel<<<(Bc * Tc) / 64, 128, 0, stream>>>(obs, w1h, w2h, w3h, b1, b2, b3,
                                                   logem);
    forward_kernel<<<Bc, 32, 0, stream>>>(logem, trans, linit, alpha, ll);
    backward_kernel<<<Bc, 32, 0, stream>>>(logem, transT, alpha, probs, marg);
}
